// ResidualMultiHeadAttention_8933531976017
// MI455X (gfx1250) — compile-verified, matrix operand order checked by reading, not run on hardware
//
#include <hip/hip_runtime.h>
#include <hip/hip_bf16.h>

// ---------------------------------------------------------------------------
// Types for CDNA5 WMMA (wave32)
// ---------------------------------------------------------------------------
typedef __bf16 bf16;
typedef bf16  v8bf  __attribute__((ext_vector_type(8)));
typedef bf16  v16bf __attribute__((ext_vector_type(16)));
typedef float v8f   __attribute__((ext_vector_type(8)));
typedef int   v4i   __attribute__((vector_size(16)));   // matches builtin param

#define NB_HEAD 16
#define DPH     64
#define EMB     1024
#define BATCH   4
#define SEQ     2048
#define MROWS   (BATCH * SEQ)      // 8192

__device__ __forceinline__ v8f wmma_bf16(v16bf a, v16bf b, v8f c) {
    // D = A(16x32) * B(32x16) + C(16x16 f32)
    return __builtin_amdgcn_wmma_f32_16x16x32_bf16(
        /*neg_a=*/false, a, /*neg_b=*/false, b,
        /*c_mod=*/(short)0, c, /*reuse_a=*/false, /*reuse_b=*/false);
}

// Load a 16x32 bf16 A/B fragment from a row-major matrix (global or LDS).
// ISA layout: lane L<16 -> row L, K = {0..7, 16..23}; lane L>=16 -> row L-16,
// K = {8..15, 24..31}. Two contiguous 16-byte loads per lane.
__device__ __forceinline__ v16bf load_frag(const bf16* base, int ld) {
    int lane = threadIdx.x & 31;
    int r    = lane & 15;
    int ko   = (lane >> 4) << 3;           // 0 or 8
    const bf16* p = base + (size_t)r * ld + ko;
    v8bf lo = *(const v8bf*)(p);
    v8bf hi = *(const v8bf*)(p + 16);
    return __builtin_shufflevector(lo, hi,
        0,1,2,3,4,5,6,7, 8,9,10,11,12,13,14,15);
}

// ---------------------------------------------------------------------------
// Direct global->LDS async copies (gfx1250 ASYNCcnt path) with sync fallback.
// The builtin wants non-const int4* in AS(1)/AS(3); we produce those via
// integer round-trips (global shares the generic address value; a generic
// shared pointer's low 32 bits are the LDS offset per the aperture mapping).
// ---------------------------------------------------------------------------
#if defined(__gfx1250__) && __has_builtin(__builtin_amdgcn_global_load_async_to_lds_b128)
#define ATHENA_ASYNC_COPY 1
#endif

__device__ __forceinline__ void copy16(bf16* lds, const bf16* g) {
#ifdef ATHENA_ASYNC_COPY
    __builtin_amdgcn_global_load_async_to_lds_b128(
        (__attribute__((address_space(1))) v4i*)(unsigned long long)(uintptr_t)g,
        (__attribute__((address_space(3))) v4i*)(unsigned int)(uintptr_t)lds,
        /*offset=*/0, /*cpol=*/0);
#else
    *(v8bf*)lds = *(const v8bf*)g;
#endif
}

__device__ __forceinline__ void wait_copies() {
#ifdef ATHENA_ASYNC_COPY
#if __has_builtin(__builtin_amdgcn_s_wait_asynccnt)
    __builtin_amdgcn_s_wait_asynccnt(0);
#else
    asm volatile("s_wait_asynccnt 0x0" ::: "memory");
#endif
#endif
    // sync fallback: DS stores are covered by the compiler's waits + barrier
}

// ---------------------------------------------------------------------------
// fp32 -> bf16 conversion, 8 elements/thread/iter (2x b128 load, 1x b128 store)
// ---------------------------------------------------------------------------
__global__ void cvt_f32_bf16_kernel(const float* __restrict__ in,
                                    bf16* __restrict__ out, size_t n8) {
    size_t i      = (size_t)blockIdx.x * blockDim.x + threadIdx.x;
    size_t stride = (size_t)gridDim.x * blockDim.x;
    for (; i < n8; i += stride) {
        const float4* p = (const float4*)(in + i * 8);
        float4 a = p[0], b = p[1];
        v8bf o;
        o[0] = (bf16)a.x; o[1] = (bf16)a.y; o[2] = (bf16)a.z; o[3] = (bf16)a.w;
        o[4] = (bf16)b.x; o[5] = (bf16)b.y; o[6] = (bf16)b.z; o[7] = (bf16)b.w;
        *(v8bf*)(out + i * 8) = o;
    }
}

// ---------------------------------------------------------------------------
// Tiled bf16 GEMM:  C[m,n] = sum_k A[m,k] * W[n,k]   (i.e. A @ W^T)
//   A: [M,K] bf16 row-major, W: [N,K] bf16 row-major.
//   Block tile 128(M)x256(N), 8 waves (2M x 4N), wave tile 64x64:
//   16 WMMA per 32-wide K step vs 16 ds_load_b128 + 1 barrier.
//   Double-buffered LDS; next tile's global->LDS copies (async path) overlap
//   the current tile's WMMAs.
// Epilogue modes:
//   0: bf16 out, [B,H,S,Dh]  (Q, K projections)
//   1: bf16 out, [B,H,Dh,S]  (V projection, transposed for PV fragments)
//   2: f32  out = acc + residual, row-major [M,N]  (output projection)
// ---------------------------------------------------------------------------
__global__ __launch_bounds__(256) void gemm_bf16_kernel(
    const bf16* __restrict__ A, const bf16* __restrict__ W,
    int M, int N, int K, void* __restrict__ out,
    const float* __restrict__ residual, int mode)
{
    __shared__ bf16 As[2][128][40];   // stride 40 halfwords = 80B: 16B aligned,
    __shared__ bf16 Ws[2][256][40];   // conflict-free for b128 fragment reads

    const int tid   = threadIdx.x;
    const int wave  = tid >> 5;
    const int lane  = tid & 31;
    const int waveM = wave & 1;          // 2 waves along M (64 rows each)
    const int waveN = wave >> 1;         // 4 waves along N (64 cols each)
    const int bm    = blockIdx.y * 128;
    const int bn    = blockIdx.x * 256;

    // Stage one 128x32 A slice + one 256x32 W slice into LDS buffer `buf`.
    auto stage = [&](int buf, int kk) {
        #pragma unroll
        for (int i = 0; i < 2; ++i) {        // A: 512 x 16B chunks
            int id = tid + i * 256;
            int row = id >> 2, cc = (id & 3) << 3;
            copy16(&As[buf][row][cc], A + (size_t)(bm + row) * K + kk + cc);
        }
        #pragma unroll
        for (int i = 0; i < 4; ++i) {        // W: 1024 x 16B chunks
            int id = tid + i * 256;
            int row = id >> 2, cc = (id & 3) << 3;
            copy16(&Ws[buf][row][cc], W + (size_t)(bn + row) * K + kk + cc);
        }
    };

    v8f acc[4][4] = {};

    stage(0, 0);
    int buf = 0;
    for (int kk = 0; kk < K; kk += 32) {
        wait_copies();          // this wave's copies into `buf` are done
        __syncthreads();        // everyone's copies visible; prev readers done
        if (kk + 32 < K)
            stage(buf ^ 1, kk + 32);   // overlaps with the WMMAs below

        v16bf bfr[4];
        #pragma unroll
        for (int nt = 0; nt < 4; ++nt)
            bfr[nt] = load_frag(&Ws[buf][waveN * 64 + nt * 16][0], 40);
        #pragma unroll
        for (int mt = 0; mt < 4; ++mt) {
            v16bf af = load_frag(&As[buf][waveM * 64 + mt * 16][0], 40);
            #pragma unroll
            for (int nt = 0; nt < 4; ++nt)
                acc[mt][nt] = wmma_bf16(af, bfr[nt], acc[mt][nt]);
        }
        buf ^= 1;
    }

    // Epilogue: C layout = lane(0..15)->N, VGPR r -> row r (+8 for lanes>=16)
    const int half = lane >> 4;
    const int col  = lane & 15;
    #pragma unroll
    for (int mt = 0; mt < 4; ++mt) {
        #pragma unroll
        for (int nt = 0; nt < 4; ++nt) {
            #pragma unroll
            for (int r = 0; r < 8; ++r) {
                int m = bm + waveM * 64 + mt * 16 + r + 8 * half;
                int n = bn + waveN * 64 + nt * 16 + col;
                float v = acc[mt][nt][r];
                if (mode == 2) {
                    size_t off = (size_t)m * N + n;
                    ((float*)out)[off] = v + residual[off];
                } else {
                    int b = m >> 11, s = m & 2047;   // SEQ = 2048
                    int h = n >> 6,  d = n & 63;     // DPH = 64
                    size_t off;
                    if (mode == 0)
                        off = (((size_t)b * NB_HEAD + h) * SEQ + s) * DPH + d;
                    else
                        off = (((size_t)b * NB_HEAD + h) * DPH + d) * SEQ + s;
                    ((bf16*)out)[off] = (bf16)v;
                }
            }
        }
    }
}

// ---------------------------------------------------------------------------
// Flash-style attention with post-softmax masking (no renormalization):
//   denom l runs over ALL keys; mask only zeroes P before the PV product.
// Block = 4 waves x 32 lanes; each wave owns 16 q-rows; block owns 64 q-rows.
// Per 32-key step: 4 WMMA (QK^T, Dh=64 contraction) + 4 WMMA (PV).
// ---------------------------------------------------------------------------
__global__ __launch_bounds__(128) void attn_kernel(
    const bf16* __restrict__ Q,          // [B,H,S,Dh]
    const bf16* __restrict__ Kmat,       // [B,H,S,Dh]
    const bf16* __restrict__ VT,         // [B,H,Dh,S]
    const unsigned char* __restrict__ mask,   // [B,Sq,Sk] bool
    bf16* __restrict__ ctx)              // [B,S,H*Dh]
{
    __shared__ bf16 pbuf[4][16][40];     // per-wave P transpose buffer

    const int tid  = threadIdx.x;
    const int wave = tid >> 5;
    const int lane = tid & 31;
    const int bid  = blockIdx.x;
    const int qt   = bid & 31;           // SEQ/64 = 32 q-tiles
    const int h    = (bid >> 5) & 15;
    const int b    = bid >> 9;
    const int q0   = qt * 64 + wave * 16;

    const bf16* Qp  = Q    + (((size_t)b * NB_HEAD + h) * SEQ + q0) * DPH;
    const bf16* Kp0 = Kmat + (((size_t)b * NB_HEAD + h) * SEQ) * DPH;
    const bf16* VTp = VT   + (((size_t)b * NB_HEAD + h) * DPH) * SEQ;
    const unsigned char* Mp = mask + (size_t)b * SEQ * SEQ;

    // Q fragments for the Dh=64 contraction (two 16x32 K-chunks)
    const v16bf aq0 = load_frag(Qp, DPH);
    const v16bf aq1 = load_frag(Qp + 32, DPH);

    const int half = lane >> 4;
    const int col  = lane & 15;

    float mrow[8], lrow[8];
    v8f o[4] = {};
    #pragma unroll
    for (int r = 0; r < 8; ++r) { mrow[r] = -3.0e38f; lrow[r] = 0.0f; }

    for (int kk = 0; kk < SEQ; kk += 32) {
        const bf16* Kp = Kp0 + (size_t)kk * DPH;

        // S = (Q K^T) / 8 for 16 q-rows x 32 keys, as two 16x16 f32 tiles
        v8f s_lo = {}, s_hi = {};
        {
            v16bf bk;
            bk = load_frag(Kp, DPH);                 s_lo = wmma_bf16(aq0, bk, s_lo);
            bk = load_frag(Kp + 32, DPH);            s_lo = wmma_bf16(aq1, bk, s_lo);
            bk = load_frag(Kp + 16 * DPH, DPH);      s_hi = wmma_bf16(aq0, bk, s_hi);
            bk = load_frag(Kp + 16 * DPH + 32, DPH); s_hi = wmma_bf16(aq1, bk, s_hi);
        }
        #pragma unroll
        for (int r = 0; r < 8; ++r) { s_lo[r] *= 0.125f; s_hi[r] *= 0.125f; }

        // Online softmax update (row reductions across the 16-lane half)
        float pl[8], ph[8];
        #pragma unroll
        for (int r = 0; r < 8; ++r) {
            float t = fmaxf(s_lo[r], s_hi[r]);
            #pragma unroll
            for (int off = 1; off < 16; off <<= 1)
                t = fmaxf(t, __shfl_xor(t, off, 32));
            float mn = fmaxf(mrow[r], t);
            float sc = __expf(mrow[r] - mn);
            mrow[r] = mn;
            float pa = __expf(s_lo[r] - mn);
            float pb = __expf(s_hi[r] - mn);
            float rs = pa + pb;
            #pragma unroll
            for (int off = 1; off < 16; off <<= 1)
                rs += __shfl_xor(rs, off, 32);
            lrow[r] = lrow[r] * sc + rs;            // denom over ALL keys
            #pragma unroll
            for (int g = 0; g < 4; ++g) o[g][r] *= sc;

            // post-softmax zero masking (True -> 0); branch-free selects
            int qrow = q0 + r + 8 * half;
            const unsigned char* mr = Mp + (size_t)qrow * SEQ + kk;
            pa = mr[col]      ? 0.0f : pa;
            pb = mr[col + 16] ? 0.0f : pb;
            pl[r] = pa; ph[r] = pb;
        }

        // Transpose P (C-layout -> A-layout) through LDS
        #pragma unroll
        for (int r = 0; r < 8; ++r) {
            pbuf[wave][r + 8 * half][col]      = (bf16)pl[r];
            pbuf[wave][r + 8 * half][col + 16] = (bf16)ph[r];
        }
        asm volatile("s_wait_dscnt 0" ::: "memory");   // same-wave LDS RAW

        v16bf ap = load_frag(&pbuf[wave][0][0], 40);
        #pragma unroll
        for (int g = 0; g < 4; ++g) {
            v16bf bv = load_frag(VTp + (size_t)(g * 16) * SEQ + kk, SEQ);
            o[g] = wmma_bf16(ap, bv, o[g]);
        }
        // DS ops from the same wave are in-order: next iteration's pbuf
        // stores (WAR vs. the ds loads above) are safe without a wait.
    }

    // Finalize: divide by the all-key denominator, emit ctx[b, q, h*64+d]
    #pragma unroll
    for (int r = 0; r < 8; ++r) {
        float inv = 1.0f / lrow[r];
        int qrow  = q0 + r + 8 * half;
        bf16* cp  = ctx + ((size_t)b * SEQ + qrow) * (NB_HEAD * DPH) + h * DPH;
        #pragma unroll
        for (int g = 0; g < 4; ++g)
            cp[g * 16 + col] = (bf16)(o[g][r] * inv);
    }
}

// ---------------------------------------------------------------------------
// Host-side orchestration
// ---------------------------------------------------------------------------
extern "C" void kernel_launch(void* const* d_in, const int* in_sizes, int n_in,
                              void* d_out, int out_size, void* d_ws, size_t ws_size,
                              hipStream_t stream) {
    (void)in_sizes; (void)n_in; (void)out_size; (void)ws_size;

    const float* Xq = (const float*)d_in[0];
    const float* Xk = (const float*)d_in[1];
    const float* Xv = (const float*)d_in[2];
    const unsigned char* mask = (const unsigned char*)d_in[3];
    const float* Wq = (const float*)d_in[4];
    const float* Wk = (const float*)d_in[5];
    const float* Wv = (const float*)d_in[6];
    const float* Wo = (const float*)d_in[7];
    float* out = (float*)d_out;

    const size_t NX = (size_t)MROWS * EMB;   // 8,388,608 activation elements
    const size_t NW = (size_t)EMB * EMB;     // 1,048,576 weight elements

    char*  ws  = (char*)d_ws;
    size_t off = 0;
    auto carve = [&](size_t bytes) -> char* {
        char* p = ws + off;
        off = (off + bytes + 255) & ~(size_t)255;
        return p;
    };
    bf16* xq = (bf16*)carve(NX * 2);
    bf16* xk = (bf16*)carve(NX * 2);
    bf16* xv = (bf16*)carve(NX * 2);
    bf16* wq = (bf16*)carve(NW * 2);
    bf16* wk = (bf16*)carve(NW * 2);
    bf16* wv = (bf16*)carve(NW * 2);
    bf16* wo = (bf16*)carve(NW * 2);
    bf16* qb = (bf16*)carve(NX * 2);   // [B,H,S,Dh]
    bf16* kb = (bf16*)carve(NX * 2);   // [B,H,S,Dh]
    bf16* vt = (bf16*)carve(NX * 2);   // [B,H,Dh,S]
    bf16* cx = (bf16*)carve(NX * 2);   // [B,S,H*Dh]

    // 1) fp32 -> bf16 (vectorized: 8 elements per thread-iteration)
    cvt_f32_bf16_kernel<<<1024, 256, 0, stream>>>(Xq, xq, NX / 8);
    cvt_f32_bf16_kernel<<<1024, 256, 0, stream>>>(Xk, xk, NX / 8);
    cvt_f32_bf16_kernel<<<1024, 256, 0, stream>>>(Xv, xv, NX / 8);
    cvt_f32_bf16_kernel<<<256, 256, 0, stream>>>(Wq, wq, NW / 8);
    cvt_f32_bf16_kernel<<<256, 256, 0, stream>>>(Wk, wk, NW / 8);
    cvt_f32_bf16_kernel<<<256, 256, 0, stream>>>(Wv, wv, NW / 8);
    cvt_f32_bf16_kernel<<<256, 256, 0, stream>>>(Wo, wo, NW / 8);

    // 2) projections (WMMA GEMMs), block tile 128x256
    dim3 gg(EMB / 256, MROWS / 128);   // (4, 64)
    gemm_bf16_kernel<<<gg, 256, 0, stream>>>(xq, wq, MROWS, EMB, EMB, qb, nullptr, 0);
    gemm_bf16_kernel<<<gg, 256, 0, stream>>>(xk, wk, MROWS, EMB, EMB, kb, nullptr, 0);
    gemm_bf16_kernel<<<gg, 256, 0, stream>>>(xv, wv, MROWS, EMB, EMB, vt, nullptr, 1);

    // 3) attention (B*H*(S/64) = 2048 blocks, 4 waves each)
    attn_kernel<<<BATCH * NB_HEAD * (SEQ / 64), 128, 0, stream>>>(qb, kb, vt, mask, cx);

    // 4) output projection + residual (fp32 out)
    gemm_bf16_kernel<<<gg, 256, 0, stream>>>(cx, wo, MROWS, EMB, EMB, out, Xq, 2);
}
